// MultiHeadAttention_39341900431503
// MI455X (gfx1250) — compile-verified, matrix operand order checked by reading, not run on hardware
//
#include <hip/hip_runtime.h>
#include <hip/hip_bf16.h>
#include <math.h>

// ---------------------------------------------------------------------------
// MHA forward (B=1, S=2048, HIDDEN=2048, H=16, D=128, fp32 io) on CDNA5.
//
// All GEMM operands converted to f16 once (halves HBM traffic; WMMA wants f16
// anyway). Every GEMM is one BT=true kernel: block 128x128, 8 waves (4Mx2N),
// wave tile 32x64 = 8 v_wmma_f32_16x16x32_f16 per K-step (K-step 32).
// Tiles are DOUBLE-BUFFERED in LDS and staged with
// GLOBAL_LOAD_ASYNC_TO_LDS_B128 (ASYNCcnt): while WMMAs consume buffer b, the
// async engine fills buffer b^1 — one barrier per K-step, no VGPR round-trip.
// The only fp32 operand is the softmax output (attn_weights must be fp32 in
// d_out); the ctx GEMM converts it during staging (traffic-neutral vs keeping
// an f16 copy, saves 128 MB of workspace).
//
// Pipeline: cvt(hidden,wqkv,wout) -> qkv gemm(f16 out) -> split+RoPE (f16
// q,k,vT + fp32 present_k/v) -> scores gemm(+mask,*scale, f32) -> softmax ->
// ctx gemm(A=f32 weights, f16 merged out) -> out-proj gemm (f32 out).
// ---------------------------------------------------------------------------

typedef __attribute__((ext_vector_type(16))) _Float16 v16h;
typedef __attribute__((ext_vector_type(8)))  float    v8f;

#define S_LEN   2048
#define HIDDEN  2048
#define HEADS   16
#define HEAD_D  128
#define QKV3    (3 * HIDDEN)
#define ATT_SCALE 0.08838834764831845f   // 128^-0.5

// LDS pitch: 72 halves = 144 B/row. Rows 16B-aligned (async B128 stores) and
// bank stagger 36*r mod 64 -> 16 distinct banks for r=0..15 fragment reads.
#define LP 72
#define TILE_HALVES (128 * LP)           // one 128x32 tile (padded)
#define TILE_BYTES  (TILE_HALVES * 2)    // 18432 B

// 16x32 f16 fragment from an LDS tile row (ISA 7.12.2 layout): two contiguous
// 8-half runs at +hi and +hi+16 -> 2x ds_load_b128.
__device__ __forceinline__ v16h load_frag(const _Float16* lds, int row, int hi)
{
    const _Float16* p = lds + row * LP + hi;
    v16h f;
#pragma unroll
    for (int x = 0; x < 8; ++x) { f[x] = p[x]; f[8 + x] = p[16 + x]; }
    return f;
}

// pack 4 floats -> 4 f16 -> one 8-byte LDS store
__device__ __forceinline__ void store_pk4(_Float16* dst, float4 f)
{
    union { _Float16 h[4]; unsigned long long u; } pk;
    pk.h[0] = (_Float16)f.x; pk.h[1] = (_Float16)f.y;
    pk.h[2] = (_Float16)f.z; pk.h[3] = (_Float16)f.w;
    *(unsigned long long*)dst = pk.u;
}

// async memory->LDS copy, 16B per lane (ISA 15.18.3 op 98, ASYNCcnt tracked)
__device__ __forceinline__ void async_to_lds_b128(unsigned lds_addr, const void* g)
{
    asm volatile("global_load_async_to_lds_b128 %0, %1, off"
                 :: "v"(lds_addr), "v"(g) : "memory");
}
__device__ __forceinline__ void wait_async0()
{
    asm volatile("s_wait_asynccnt 0x0" ::: "memory");
}

// ---------------------------------------------------------------------------
// WMMA GEMM:  C[z] = (A[z](MxK,row) * B[z](NxK,row)^T + mask) * scale
//   AF16   : A is f16 (async staging) else f32 (convert during staging)
//   OUTF16 : C stored as f16 (no mask/scale) else f32
//   MASKED : fp32 epilogue adds mask[row,col] before scaling
// Requires M%128==0, N%128==0, K%64==0 (>=2 K-steps), 16B-aligned rows.
// ---------------------------------------------------------------------------
template <bool AF16, bool OUTF16, bool MASKED>
__global__ __launch_bounds__(256) void wmma_gemm_kernel(
    const void* __restrict__ Av, const _Float16* __restrict__ B,
    void* __restrict__ Cv, const float* __restrict__ mask, float scale,
    int K, int lda, int ldb, int ldc, int cstride, int mask_ld,
    long long aStride, long long bStride, long long cStride)
{
    __shared__ __align__(16) _Float16 lds_a[2][TILE_HALVES];
    __shared__ __align__(16) _Float16 lds_b[2][TILE_HALVES];

    const int z = blockIdx.z;
    const int nBase = blockIdx.x * 128;
    const int mBase = blockIdx.y * 128;
    const int t     = threadIdx.x;
    const int lane  = t & 31;
    const int wave  = t >> 5;
    const int wr    = (wave & 3) * 32;   // wave M offset in block tile
    const int wc    = (wave >> 2) * 64;  // wave N offset in block tile
    const int r     = lane & 15;
    const int hi    = (lane & 16) ? 8 : 0;

    const _Float16* Bh = B + z * bStride;
    const unsigned lza = (unsigned)(unsigned long long)&lds_a[0][0];
    const unsigned lzb = (unsigned)(unsigned long long)&lds_b[0][0];

    // per-lane async copy geometry: 8 halves (16B) per lane, 8 rows per inst
    const int cRow = lane >> 2;          // 0..7
    const int cKo  = (lane & 3) * 8;     // 0,8,16,24 halves

    // stage one 128x32 K-slice into LDS buffer `buf`
    auto stage = [&](int kt, int buf) {
        const unsigned bofs = (unsigned)buf * TILE_BYTES;
#pragma unroll
        for (int ii = 0; ii < 2; ++ii) {               // B tile: async f16
            int n = wave * 16 + ii * 8 + cRow;
            async_to_lds_b128(lzb + bofs + (unsigned)(n * LP + cKo) * 2,
                              Bh + (long long)(nBase + n) * ldb + kt + cKo);
        }
        if (AF16) {                                    // A tile: async f16
            const _Float16* Ah = (const _Float16*)Av + z * aStride;
#pragma unroll
            for (int ii = 0; ii < 2; ++ii) {
                int row = wave * 16 + ii * 8 + cRow;
                async_to_lds_b128(lza + bofs + (unsigned)(row * LP + cKo) * 2,
                                  Ah + (long long)(mBase + row) * lda + kt + cKo);
            }
        } else {                                       // A tile: f32 -> f16
            const float* Af = (const float*)Av + z * aStride;
#pragma unroll
            for (int i = 0; i < 4; ++i) {
                int e   = i * 256 + t;
                int row = e >> 3;                      // 8 float4 per row
                int k4  = (e & 7) * 4;
                float4 f = *(const float4*)(Af + (long long)(mBase + row) * lda + kt + k4);
                store_pk4(&lds_a[buf][row * LP + k4], f);
            }
        }
    };

    v8f acc[2][4];
#pragma unroll
    for (int i = 0; i < 2; ++i)
#pragma unroll
        for (int j = 0; j < 4; ++j) acc[i][j] = (v8f){};

    stage(0, 0);
    int buf = 0;
    for (int kt = 0; kt < K; kt += 32) {
        wait_async0();          // this wave's pending tile transfers done
        __syncthreads();        // tile `buf` visible; buf^1 free to refill
        if (kt + 32 < K) stage(kt + 32, buf ^ 1);   // overlaps with WMMAs

        const _Float16* la = lds_a[buf];
        const _Float16* lb = lds_b[buf];
        v16h a0 = load_frag(la, wr + r, hi);
        v16h a1 = load_frag(la, wr + 16 + r, hi);
        v16h b[4];
#pragma unroll
        for (int j = 0; j < 4; ++j) b[j] = load_frag(lb, wc + 16 * j + r, hi);
#pragma unroll
        for (int j = 0; j < 4; ++j)
            acc[0][j] = __builtin_amdgcn_wmma_f32_16x16x32_f16(
                false, a0, false, b[j], (short)0, acc[0][j], false, false);
#pragma unroll
        for (int j = 0; j < 4; ++j)
            acc[1][j] = __builtin_amdgcn_wmma_f32_16x16x32_f16(
                false, a1, false, b[j], (short)0, acc[1][j], false, false);
        buf ^= 1;
    }

    // ---- epilogue: f32 C/D layout: vgpr rr -> row rr (+8 lanes>=16) ----
#pragma unroll
    for (int i = 0; i < 2; ++i) {
        const int row0 = mBase + wr + i * 16 + hi;
#pragma unroll
        for (int j = 0; j < 4; ++j) {
            const int col = nBase + wc + j * 16 + r;
#pragma unroll
            for (int rr = 0; rr < 8; ++rr) {
                int row = row0 + rr;
                long long ci = (long long)row * ldc + (long long)col * cstride
                             + z * cStride;
                if (OUTF16) {
                    ((_Float16*)Cv)[ci] = (_Float16)acc[i][j][rr];
                } else {
                    float m = MASKED ? mask[(long long)row * mask_ld + col] : 0.0f;
                    ((float*)Cv)[ci] = (acc[i][j][rr] + m) * scale;
                }
            }
        }
    }
}

// ---------------------------------------------------------------------------
// fp32 -> f16 conversion, 4 elements/thread, packed 8B stores
// ---------------------------------------------------------------------------
__global__ __launch_bounds__(256) void cvt_f16_kernel(
    const float* __restrict__ src, _Float16* __restrict__ dst)
{
    long long i4 = ((long long)blockIdx.x * 256 + threadIdx.x) * 4;
    float4 f = *(const float4*)(src + i4);
    union { _Float16 h[4]; unsigned long long u; } pk;
    pk.h[0] = (_Float16)f.x; pk.h[1] = (_Float16)f.y;
    pk.h[2] = (_Float16)f.z; pk.h[3] = (_Float16)f.w;
    *(unsigned long long*)(dst + i4) = pk.u;
}

// ---------------------------------------------------------------------------
// Split f16 qkv (S x 6144, col = d*HEADS + h) into per-head tensors; RoPE on
// q,k (fp32 math); emit f16 q,k (H,S,D), f16 vT (H,D,S) for the BT=true ctx
// GEMM, and fp32 present_k/v (pre-RoPE) to d_out.
// ---------------------------------------------------------------------------
__global__ __launch_bounds__(256) void split_rope_kernel(
    const _Float16* __restrict__ qkv,
    _Float16* __restrict__ q_h, _Float16* __restrict__ k_h,
    _Float16* __restrict__ vT,
    float* __restrict__ present_k, float* __restrict__ present_v)
{
    int idx = blockIdx.x * 4 + (threadIdx.x >> 6);   // (h,s) id in [0, H*S)
    int j   = threadIdx.x & 63;                      // freq index 0..63
    int h   = idx >> 11;                             // / S_LEN
    int s   = idx & (S_LEN - 1);

    const _Float16* row = qkv + (long long)s * QKV3;
    int cLo = j * HEADS + h;
    int cHi = (j + 64) * HEADS + h;

    float q1 = (float)row[cLo],              q2 = (float)row[cHi];
    float k1 = (float)row[HIDDEN + cLo],     k2 = (float)row[HIDDEN + cHi];
    float v1 = (float)row[2 * HIDDEN + cLo], v2 = (float)row[2 * HIDDEN + cHi];

    float inv_freq = powf(10000.0f, -(2.0f * (float)j) / (float)HEAD_D);
    float ang = (float)s * inv_freq;
    float cs = cosf(ang), sn = sinf(ang);

    long long o = ((long long)h * S_LEN + s) * HEAD_D;
    q_h[o + j]      = (_Float16)(q1 * cs - q2 * sn);
    q_h[o + j + 64] = (_Float16)(q2 * cs + q1 * sn);
    k_h[o + j]      = (_Float16)(k1 * cs - k2 * sn);
    k_h[o + j + 64] = (_Float16)(k2 * cs + k1 * sn);
    present_k[o + j]      = k1;  present_k[o + j + 64] = k2;
    present_v[o + j]      = v1;  present_v[o + j + 64] = v2;
    // vT[h][d][s]
    long long vb = (long long)h * HEAD_D * S_LEN + s;
    vT[vb + (long long)j * S_LEN]        = (_Float16)v1;
    vT[vb + (long long)(j + 64) * S_LEN] = (_Float16)v2;
}

// ---------------------------------------------------------------------------
// Row softmax in place over fp32 attn_weights: one block per 2048-long row.
// ---------------------------------------------------------------------------
__global__ __launch_bounds__(256) void softmax_kernel(float* __restrict__ w)
{
    float* p = w + (long long)blockIdx.x * S_LEN;
    int t = threadIdx.x;
    __shared__ float red[256];

    float vals[8];
    float mx = -INFINITY;
#pragma unroll
    for (int i = 0; i < 8; ++i) { vals[i] = p[i * 256 + t]; mx = fmaxf(mx, vals[i]); }
    red[t] = mx; __syncthreads();
    for (int sft = 128; sft > 0; sft >>= 1) {
        if (t < sft) red[t] = fmaxf(red[t], red[t + sft]);
        __syncthreads();
    }
    mx = red[0]; __syncthreads();

    float sum = 0.0f;
#pragma unroll
    for (int i = 0; i < 8; ++i) { vals[i] = __expf(vals[i] - mx); sum += vals[i]; }
    red[t] = sum; __syncthreads();
    for (int sft = 128; sft > 0; sft >>= 1) {
        if (t < sft) red[t] += red[t + sft];
        __syncthreads();
    }
    float inv = 1.0f / red[0];
#pragma unroll
    for (int i = 0; i < 8; ++i) p[i * 256 + t] = vals[i] * inv;
}

// ---------------------------------------------------------------------------
extern "C" void kernel_launch(void* const* d_in, const int* in_sizes, int n_in,
                              void* d_out, int out_size, void* d_ws, size_t ws_size,
                              hipStream_t stream)
{
    const float* hidden = (const float*)d_in[0];   // (1, 2048, 2048)
    const float* mask   = (const float*)d_in[1];   // (2048, 2048)
    const float* w_qkv  = (const float*)d_in[2];   // (6144, 2048)
    const float* w_out  = (const float*)d_in[3];   // (2048, 2048)

    const long long SD = (long long)S_LEN * HEAD_D;        // 262144
    const long long SS = (long long)S_LEN * S_LEN;         // 4M
    const long long SH = (long long)S_LEN * HIDDEN;        // 4M

    // d_out: attn_output | attn_weights | present_k | present_v (flat, fp32)
    float* out_attn = (float*)d_out;                    // 4M
    float* weights  = out_attn + SH;                    // 64M
    float* pres_k   = weights + (long long)HEADS * SS;  // 4M
    float* pres_v   = pres_k + (long long)HEADS * SD;   // 4M

    // f16 workspace (all regions 8MB-aligned), total ~96 MB
    _Float16* hidden_h = (_Float16*)d_ws;                 // 4M halves
    _Float16* wqkv_h   = hidden_h + SH;                   // 12M
    _Float16* wout_h   = wqkv_h + 3 * SH;                 // 4M
    _Float16* qkv_h    = wout_h + SH;                     // 12M
    _Float16* q_h      = qkv_h + 3 * SH;                  // 4M
    _Float16* k_h      = q_h + SH;                        // 4M
    _Float16* vT_h     = k_h + SH;                        // 4M
    _Float16* ctxm_h   = vT_h + SH;                       // 4M

    dim3 blk(256);

    // 0) one-shot fp32 -> f16 operand conversions
    cvt_f16_kernel<<<dim3((unsigned)(SH / 1024)), blk, 0, stream>>>(hidden, hidden_h);
    cvt_f16_kernel<<<dim3((unsigned)(3 * SH / 1024)), blk, 0, stream>>>(w_qkv, wqkv_h);
    cvt_f16_kernel<<<dim3((unsigned)(SH / 1024)), blk, 0, stream>>>(w_out, wout_h);

    // 1) qkv = hidden @ w_qkv^T  (2048 x 6144, K=2048), f16 out
    wmma_gemm_kernel<true, true, false>
        <<<dim3(QKV3 / 128, S_LEN / 128, 1), blk, 0, stream>>>(
        hidden_h, wqkv_h, qkv_h, nullptr, 1.0f,
        HIDDEN, HIDDEN, HIDDEN, QKV3, 1, 0, 0, 0, 0);

    // 2) split heads + RoPE + present_k/v + vT
    split_rope_kernel<<<dim3(HEADS * S_LEN / 4), blk, 0, stream>>>(
        qkv_h, q_h, k_h, vT_h, pres_k, pres_v);

    // 3) scores[h] = (mask + q[h] @ k[h]^T) * SCALE, fp32 out (z = heads)
    wmma_gemm_kernel<true, false, true>
        <<<dim3(S_LEN / 128, S_LEN / 128, HEADS), blk, 0, stream>>>(
        q_h, k_h, weights, mask, ATT_SCALE,
        HEAD_D, HEAD_D, HEAD_D, S_LEN, 1, S_LEN, SD, SD, SS);

    // 4) softmax rows in place (H*S rows)
    softmax_kernel<<<dim3(HEADS * S_LEN), blk, 0, stream>>>(weights);

    // 5) ctx[h] = weights[h] @ vT[h]^T ; A fp32 (convert in staging),
    //    f16 merged output col = d*HEADS + h
    wmma_gemm_kernel<false, true, false>
        <<<dim3(HEAD_D / 128, S_LEN / 128, HEADS), blk, 0, stream>>>(
        weights, vT_h, ctxm_h, nullptr, 1.0f,
        S_LEN, S_LEN, S_LEN, HIDDEN, HEADS, 0, SS, SD, 1);

    // 6) attn_output = ctxm @ w_out^T  (2048 x 2048, K=2048), fp32 out
    wmma_gemm_kernel<true, false, false>
        <<<dim3(HIDDEN / 128, S_LEN / 128, 1), blk, 0, stream>>>(
        ctxm_h, wout_h, out_attn, nullptr, 1.0f,
        HIDDEN, HIDDEN, HIDDEN, HIDDEN, 1, 0, 0, 0, 0);
}
